// MarginLoss_6124623364216
// MI455X (gfx1250) — compile-verified, matrix operand order checked by reading, not run on hardware
//
#include <hip/hip_runtime.h>
#include <hip/hip_bf16.h>
#include <math.h>

typedef __attribute__((ext_vector_type(16))) __bf16 v16bf;
typedef __attribute__((ext_vector_type(8)))  float  v8f;

#define BATCH 4096
#define DIM 128
#define MARGIN 0.2f
#define NEG_INF (-1e30f)

// Raw v_log_f32 (log2). Inputs are guaranteed normal floats, so we skip the
// libm denorm fixups. TRANS32 result hazard guarded like compiler codegen does.
__device__ __forceinline__ float rlog2(float x) {
  float r;
  asm volatile("v_log_f32_e32 %0, %1\n\t"
               "s_delay_alu instid0(TRANS32_DEP_1)"
               : "=v"(r) : "v"(x));
  return r;
}

// Cheap 32-bit mixer (2 muls). (i,j) is 24 bits -> full avalanche is plenty.
__device__ __forceinline__ unsigned hash32(unsigned x) {
  x ^= x >> 16; x *= 0x7FEB352Du;
  x ^= x >> 15; x *= 0x846CA68Bu;
  x ^= x >> 16;
  return x;
}

// IEEE-754 order-preserving float -> u32 map (monotone: a<b <=> key(a)<key(b)).
__device__ __forceinline__ unsigned sortable(float f) {
  unsigned s = __float_as_uint(f);
  return s ^ ((unsigned)((int)s >> 31) | 0x80000000u);
}

// Convert embeddings f32 -> bf16 (packed stores), per-row squared norms, zero accum.
__global__ void __launch_bounds__(256)
prep_kernel(const float* __restrict__ emb, __bf16* __restrict__ embh,
            float* __restrict__ sq, float* __restrict__ accum) {
  if (blockIdx.x == 0 && threadIdx.x < 2) accum[threadIdx.x] = 0.0f;
  const int row  = blockIdx.x * 8 + (threadIdx.x >> 5);
  const int lane = threadIdx.x & 31;
  if (row >= BATCH) return;
  const float4 v = ((const float4*)(emb + (size_t)row * DIM))[lane];
  float s = v.x * v.x + v.y * v.y + v.z * v.z + v.w * v.w;
  union { __bf16 h[4]; uint2 u; } pk;
  pk.h[0] = (__bf16)v.x; pk.h[1] = (__bf16)v.y; pk.h[2] = (__bf16)v.z; pk.h[3] = (__bf16)v.w;
  ((uint2*)(embh + (size_t)row * DIM))[lane] = pk.u;
#pragma unroll
  for (int m = 16; m >= 1; m >>= 1) s += __shfl_xor(s, m, 32);
  if (lane == 0) sq[row] = s;
}

// Fused Gram-tile (bf16 WMMA) + distance-weighted triplet sampling.
// Packed-key argmax: key = (sortable(value) & 0xFFFFF000) | column (12 bits),
// so a running argmax is a single umax and index is recovered with & 0xFFF.
__global__ void __launch_bounds__(256)
sample_kernel(const __bf16* __restrict__ embh, const float* __restrict__ sq,
              const int* __restrict__ labels,
              int* __restrict__ pos_out, int* __restrict__ neg_out) {
  __shared__ __bf16 sA[16 * DIM];
  __shared__ float sSq[16];
  __shared__ int   sLab[16];
  __shared__ unsigned sN[8][16];
  __shared__ unsigned sP[8][16];

  const int tid  = threadIdx.x;
  const int wv   = tid >> 5;
  const int lane = tid & 31;
  const int half = lane >> 4;
  const int l16  = lane & 15;
  const int rowbase = blockIdx.x * 16;

  {  // strip fill: 2048 bf16 = 256 uint4, one per thread
    const uint4* srcA = (const uint4*)(embh + (size_t)rowbase * DIM);
    ((uint4*)sA)[tid] = srcA[tid];
  }
  if (tid < 16) { sSq[tid] = sq[rowbase + tid]; sLab[tid] = labels[rowbase + tid]; }
  __syncthreads();

  // Hoist loop-invariant per-row values out of LDS into registers.
  float sqRow[8]; int labRow[8];
#pragma unroll
  for (int v = 0; v < 8; ++v) { sqRow[v] = sSq[v + 8 * half]; labRow[v] = sLab[v + 8 * half]; }

  // A fragments (K chunks of 32) per the 16-bit A layout.
  const unsigned* sA32 = (const unsigned*)sA;
  union VB { v16bf h; unsigned u[8]; };
  VB aF[4];
#pragma unroll
  for (int kc = 0; kc < 4; ++kc)
#pragma unroll
    for (int v = 0; v < 8; ++v)
      aF[kc].u[v] = sA32[l16 * 64 + kc * 16 + ((v >= 4) ? 8 : 0) + half * 4 + (v & 3)];

  unsigned runN[8], runP[8];
#pragma unroll
  for (int v = 0; v < 8; ++v) { runN[v] = 0u; runP[v] = 0u; }

  for (int ct = wv; ct < BATCH / 16; ct += 8) {
    const int colrow = ct * 16 + l16;              // this lane's column index j
    const unsigned* bp32 = (const unsigned*)(embh + (size_t)colrow * DIM);
    const float sqj = sq[colrow];
    const int   labj = labels[colrow];
    // branchless prefetch of the tile 1 iteration (8 tiles) ahead (wraps; always valid)
    __builtin_prefetch(embh + (size_t)((colrow + 128) & (BATCH - 1)) * DIM, 0, 3);

    VB bF[4];
#pragma unroll
    for (int kc = 0; kc < 4; ++kc)
#pragma unroll
      for (int v = 0; v < 8; ++v)
        bF[kc].u[v] = bp32[kc * 16 + ((v >= 4) ? 8 : 0) + half * 4 + (v & 3)];

    v8f acc = {0.f, 0.f, 0.f, 0.f, 0.f, 0.f, 0.f, 0.f};
#pragma unroll
    for (int kc = 0; kc < 4; ++kc)
      acc = __builtin_amdgcn_wmma_f32_16x16x32_bf16(false, aF[kc].h, false, bF[kc].h,
                                                    (short)0, acc, false, false);

    // Branchless epilogue in d^2/log2 space. argmax invariant under common
    // positive scale + shift:
    //   cn' = -(63*log2(d2c) + 62.5*log2(1-0.25*d2c) + log2(-log2(u)))
    // One hash per (i,j): neg (diff-label) and pos (same-label) masks are disjoint.
#pragma unroll
    for (int v = 0; v < 8; ++v) {
      const int gi = rowbase + v + 8 * half;
      float d2  = fmaxf(sqRow[v] + sqj - 2.0f * acc[v], 0.0f);
      float d2c = fmaxf(d2, 0.25f);
      float l2d = rlog2(d2c);
      float l2w = rlog2(fmaxf(1.0f - 0.25f * d2c, 1e-20f));
      unsigned h = hash32((((unsigned)gi << 12) | (unsigned)colrow) ^ 0x9E3779B9u) | 1u;
      float u   = fminf((float)h * 2.3283064e-10f, 0.99999994f);   // (0,1)
      float l2t = rlog2(-rlog2(u));
      float cn  = -(63.0f * l2d + 62.5f * l2w + l2t);
      const bool same = (labRow[v] == labj);

      unsigned nk = (sortable(same ? NEG_INF : cn) & 0xFFFFF000u) | (unsigned)colrow;
      runN[v] = max(runN[v], nk);

      unsigned pk = (same && (gi != colrow)) ? ((h & 0xFFFFF000u) | (unsigned)colrow) : 0u;
      runP[v] = max(runP[v], pk);
    }
  }

  // One-time cross-lane (16-lane group) umax per row; stage per-wave results.
#pragma unroll
  for (int v = 0; v < 8; ++v) {
    unsigned nk = runN[v], pk = runP[v];
#pragma unroll
    for (int m = 8; m >= 1; m >>= 1) {
      nk = max(nk, (unsigned)__shfl_xor((int)nk, m, 16));
      pk = max(pk, (unsigned)__shfl_xor((int)pk, m, 16));
    }
    if (l16 == 0) {
      int r = v + 8 * half;
      sN[wv][r] = nk;
      sP[wv][r] = pk;
    }
  }
  __syncthreads();
  if (tid < 16) {
    unsigned nk = 0u, pk = 0u;
    for (int w = 0; w < 8; ++w) {
      nk = max(nk, sN[w][tid]);
      pk = max(pk, sP[w][tid]);
    }
    neg_out[rowbase + tid] = (int)(nk & 0xFFFu);
    pos_out[rowbase + tid] = (int)(pk & 0xFFFu);
  }
}

// Triplet margin loss over sampled (a,p,n); f32 distances; one wave per anchor.
__global__ void __launch_bounds__(256)
loss_kernel(const float* __restrict__ emb, const int* __restrict__ labels,
            const float* __restrict__ beta, const int* __restrict__ pos_idx,
            const int* __restrict__ neg_idx, float* __restrict__ accum) {
  const int anchor = blockIdx.x * 8 + (threadIdx.x >> 5);
  const int lane = threadIdx.x & 31;
  if (anchor >= BATCH) return;
  const float* ei = emb + (size_t)anchor * DIM;
  const float* ep = emb + (size_t)pos_idx[anchor] * DIM;
  const float* en = emb + (size_t)neg_idx[anchor] * DIM;
  float sp = 0.f, sn = 0.f;
#pragma unroll
  for (int t = 0; t < 4; ++t) {
    int k = lane * 4 + t;
    float dp = ei[k] - ep[k]; sp += dp * dp;
    float dn = ei[k] - en[k]; sn += dn * dn;
  }
#pragma unroll
  for (int m = 16; m >= 1; m >>= 1) { sp += __shfl_xor(sp, m, 32); sn += __shfl_xor(sn, m, 32); }
  if (lane == 0) {
    float d_ap = sqrtf(sp + 1e-8f), d_an = sqrtf(sn + 1e-8f);
    float ba = beta[labels[anchor]];
    float pl = fmaxf(d_ap - ba + MARGIN, 0.f);
    float nl = fmaxf(ba - d_an + MARGIN, 0.f);
    float cnt = (pl > 0.f ? 1.f : 0.f) + (nl > 0.f ? 1.f : 0.f);
    atomicAdd(&accum[0], pl + nl);
    atomicAdd(&accum[1], cnt);
  }
}

__global__ void finalize_kernel(const float* __restrict__ accum, float* __restrict__ out) {
  float total = accum[0], cnt = accum[1];
  out[0] = (cnt == 0.0f) ? total : total / cnt;
}

extern "C" void kernel_launch(void* const* d_in, const int* in_sizes, int n_in,
                              void* d_out, int out_size, void* d_ws, size_t ws_size,
                              hipStream_t stream) {
  const float* emb   = (const float*)d_in[0];   // [4096,128] f32
  const int* labels  = (const int*)d_in[1];     // [4096] i32
  const float* beta  = (const float*)d_in[2];   // [100] f32

  char* ws = (char*)d_ws;
  __bf16* embh = (__bf16*)(ws);                           // 1,048,576 B
  float*  sqn  = (float*)(ws + 1048576);                  //    16,384 B
  int*    pos  = (int*)(ws + 1048576 + 16384);            //    16,384 B
  int*    neg  = (int*)(ws + 1048576 + 2 * 16384);        //    16,384 B
  float*  acc  = (float*)(ws + 1048576 + 3 * 16384);      //         8 B

  prep_kernel<<<BATCH / 8, 256, 0, stream>>>(emb, embh, sqn, acc);
  sample_kernel<<<BATCH / 16, 256, 0, stream>>>(embh, sqn, labels, pos, neg);
  loss_kernel<<<BATCH / 8, 256, 0, stream>>>(emb, labels, beta, pos, neg, acc);
  finalize_kernel<<<1, 1, 0, stream>>>(acc, (float*)d_out);
}
